// Encoder_69217692942600
// MI455X (gfx1250) — matrix operand order repaired and checked by reading, not compiled, not run on hardware
//
#include <hip/hip_runtime.h>
#include <hip/hip_bf16.h>

#define BN_EPS 1e-5f

typedef __attribute__((ext_vector_type(16))) _Float16 v16h;
typedef __attribute__((ext_vector_type(8)))  float    v8f;

// ---------------------------------------------------------------------------
// CDNA5 async copy: 16B global -> LDS via GLOBAL_LOAD_ASYNC_TO_LDS_B128
// (tracked by ASYNCcnt).  LDS operand VGPR = wave-relative LDS byte address
// = low 32 bits of the generic pointer (ISA 10.2 aperture truncation).
// ---------------------------------------------------------------------------
__device__ __forceinline__ void async_copy_b128(const void* g, void* lds) {
  unsigned           l  = (unsigned)(unsigned long long)lds;
  unsigned long long ga = (unsigned long long)g;
  asm volatile("global_load_async_to_lds_b128 %0, %1, off"
               :: "v"(l), "v"(ga) : "memory");
}
__device__ __forceinline__ void async_wait0() {
  asm volatile("s_wait_asynccnt 0" ::: "memory");
}

// ---------------------------------------------------------------------------
// WMMA fragment loaders (CDNA5 16x16x32 f16, wave32). Layouts per ISA 7.12.2.
// ---------------------------------------------------------------------------
__device__ __forceinline__ v16h frag_a(const _Float16* As, int row, int ldA) {
  const int l  = threadIdx.x & 31;
  const int m  = l & 15;
  const int kb = (l < 16) ? 0 : 8;
  const _Float16* p = As + (row + m) * ldA;
  v16h a;
#pragma unroll
  for (int e = 0; e < 8; ++e) { a[e] = p[kb + e]; a[e + 8] = p[16 + kb + e]; }
  return a;
}

__device__ __forceinline__ v16h frag_b(const _Float16* Bs, int col, int ldB) {
  const int l = threadIdx.x & 31;
  const int kb = (l < 16) ? 0 : 8;
  const _Float16* p = Bs + col + (l & 15);
  v16h b;
#pragma unroll
  for (int e = 0; e < 8; ++e) { b[e] = p[(kb + e) * ldB]; b[e + 8] = p[(16 + kb + e) * ldB]; }
  return b;
}

// ---------------------------------------------------------------------------
// Shared-weight linear:  Y = act( BN( X @ W^T + bias ) [+ res] )
// X:[M,K] f32, W:[N,K] f32, Y:[M,N] f32. act: 0 none, 1 relu, 2 lrelu.
// Block = 128 thr (4 waves).  Block tile 64x64; each wave owns a 32x32
// sub-tile = 2x2 WMMA accumulators (4 v_wmma per 32-wide K step).
// ---------------------------------------------------------------------------
__global__ __launch_bounds__(128)
void k_gemm_lin(const float* __restrict__ X, const float* __restrict__ W,
                const float* __restrict__ bias,
                const float* __restrict__ gam, const float* __restrict__ bet,
                const float* __restrict__ mea, const float* __restrict__ var,
                const float* __restrict__ res,
                float* __restrict__ Y, int M, int N, int K, int act)
{
  __shared__ _Float16 As[64 * 40];   // 64 rows x 32 k   (row stride 80B, 16B-aligned)
  __shared__ _Float16 Bs[32 * 72];   // 32 k   x 64 n    (row stride 144B, 16B-aligned)
  const int tid  = threadIdx.x;
  const int row0 = blockIdx.x * 64;
  const int col0 = blockIdx.y * 64;
  const int wave = tid >> 5;
  const int wr = (wave >> 1) * 32;   // wave row offset in block tile
  const int wc = (wave & 1) * 32;    // wave col offset in block tile
  v8f acc00 = {}, acc01 = {}, acc10 = {}, acc11 = {};

  for (int k0 = 0; k0 < K; k0 += 32) {
    // stage A: 64x32, f32 -> f16.  thread: row = tid/2, 16 contiguous cols.
    {
      const int r  = tid >> 1;
      const int cb = (tid & 1) << 4;
      const float* xp = X + (long)(row0 + r) * K + k0 + cb;
      _Float16* ap = &As[r * 40 + cb];
#pragma unroll
      for (int i = 0; i < 16; ++i) ap[i] = (_Float16)xp[i];
    }
    // stage B: Bs[k][n] = W[col0+n][k0+k].  thread: k = tid/4, 16 n's.
    {
      const int k  = tid >> 2;
      const int nb = (tid & 3) << 4;
      _Float16* bp = &Bs[k * 72 + nb];
#pragma unroll
      for (int i = 0; i < 16; ++i)
        bp[i] = (_Float16)W[(long)(col0 + nb + i) * K + k0 + k];
    }
    __syncthreads();

    v16h a0 = frag_a(As, wr,      40);
    v16h a1 = frag_a(As, wr + 16, 40);
    v16h b0 = frag_b(Bs, wc,      72);
    v16h b1 = frag_b(Bs, wc + 16, 72);
    acc00 = __builtin_amdgcn_wmma_f32_16x16x32_f16(false, a0, false, b0, (short)0, acc00, false, false);
    acc01 = __builtin_amdgcn_wmma_f32_16x16x32_f16(false, a0, false, b1, (short)0, acc01, false, false);
    acc10 = __builtin_amdgcn_wmma_f32_16x16x32_f16(false, a1, false, b0, (short)0, acc10, false, false);
    acc11 = __builtin_amdgcn_wmma_f32_16x16x32_f16(false, a1, false, b1, (short)0, acc11, false, false);
    __syncthreads();
  }

  // epilogue: C layout lane l -> n = l%16, element e -> m = e + (l<16?0:8)
  const int l  = tid & 31;
  const int nn = l & 15;
  const int mb = (l < 16) ? 0 : 8;
#pragma unroll
  for (int j = 0; j < 2; ++j) {
    const int n = col0 + wc + j * 16 + nn;
    const float s  = gam[n] * rsqrtf(var[n] + BN_EPS);
    const float sh = (bias[n] - mea[n]) * s + bet[n];
#pragma unroll
    for (int i = 0; i < 2; ++i) {
      const v8f* accp = j ? (i ? &acc11 : &acc01) : (i ? &acc10 : &acc00);
#pragma unroll
      for (int e = 0; e < 8; ++e) {
        const int m = row0 + wr + i * 16 + mb + e;
        float v = (*accp)[e] * s + sh;
        if (res) v += res[(long)m * N + n];
        if (act == 1)      v = v > 0.f ? v : 0.f;
        else if (act == 2) v = v >= 0.f ? v : 0.01f * v;
        Y[(long)m * N + n] = v;
      }
    }
  }
}

// ---------------------------------------------------------------------------
// Batched adjacency aggregation: Y[b] = amh[b] (512x512 f16) @ X[b] (512xC f32)
// grid = (512/64, C/64, B).  A-tile staged with async-to-LDS (pure f16 copy).
// ---------------------------------------------------------------------------
__global__ __launch_bounds__(128)
void k_gemm_agg(const _Float16* __restrict__ A, const float* __restrict__ X,
                float* __restrict__ Y, int C)
{
  __shared__ _Float16 As[64 * 40];
  __shared__ _Float16 Bs[32 * 72];
  const int tid  = threadIdx.x;
  const int b    = blockIdx.z;
  const int row0 = blockIdx.x * 64;
  const int col0 = blockIdx.y * 64;
  const int wave = tid >> 5;
  const int wr = (wave >> 1) * 32;
  const int wc = (wave & 1) * 32;
  const _Float16* Ab = A + (long)b * 512 * 512;
  const float*    Xb = X + (long)b * 512 * C;
  v8f acc00 = {}, acc01 = {}, acc10 = {}, acc11 = {};

  for (int k0 = 0; k0 < 512; k0 += 32) {
    // stage A (f16 copy, 32B per thread) via async DMA to LDS
    {
      const int r  = tid >> 1;
      const int kb = (tid & 1) << 4;
      const _Float16* gp = Ab + (long)(row0 + r) * 512 + k0 + kb;
      _Float16* lp = &As[r * 40 + kb];
      async_copy_b128(gp,     lp);
      async_copy_b128(gp + 8, lp + 8);
    }
    // stage B: f32 -> f16 convert, Bs[k][n] = X[k0+k][col0+n]
    {
      const int k  = tid >> 2;
      const int nb = (tid & 3) << 4;
      const float* xp = Xb + (long)(k0 + k) * C + col0 + nb;
      _Float16* bp = &Bs[k * 72 + nb];
#pragma unroll
      for (int i = 0; i < 16; ++i) bp[i] = (_Float16)xp[i];
    }
    async_wait0();
    __syncthreads();

    v16h a0 = frag_a(As, wr,      40);
    v16h a1 = frag_a(As, wr + 16, 40);
    v16h b0 = frag_b(Bs, wc,      72);
    v16h b1 = frag_b(Bs, wc + 16, 72);
    acc00 = __builtin_amdgcn_wmma_f32_16x16x32_f16(false, a0, false, b0, (short)0, acc00, false, false);
    acc01 = __builtin_amdgcn_wmma_f32_16x16x32_f16(false, a0, false, b1, (short)0, acc01, false, false);
    acc10 = __builtin_amdgcn_wmma_f32_16x16x32_f16(false, a1, false, b0, (short)0, acc10, false, false);
    acc11 = __builtin_amdgcn_wmma_f32_16x16x32_f16(false, a1, false, b1, (short)0, acc11, false, false);
    __syncthreads();
  }

  const int l  = tid & 31;
  const int nn = l & 15;
  const int mb = (l < 16) ? 0 : 8;
#pragma unroll
  for (int j = 0; j < 2; ++j) {
    const int n = col0 + wc + j * 16 + nn;
#pragma unroll
    for (int i = 0; i < 2; ++i) {
      const v8f* accp = j ? (i ? &acc11 : &acc01) : (i ? &acc10 : &acc00);
#pragma unroll
      for (int e = 0; e < 8; ++e) {
        const int m = row0 + wr + i * 16 + mb + e;
        Y[((long)b * 512 + m) * C + n] = (*accp)[e];
      }
    }
  }
}

// ---------------------------------------------------------------------------
// Masked f16 adjacency:  amh[b,i,j] = (i<ld && j<ld) ? (f16)am : 0
// ---------------------------------------------------------------------------
__global__ void k_amh(const float* __restrict__ am, const int* __restrict__ ld,
                      _Float16* __restrict__ amh)
{
  const int idx = blockIdx.x * 256 + threadIdx.x;       // B*512*512 = 8388608
  const int b = idx >> 18;
  const int r = (idx >> 9) & 511;
  const int c = idx & 511;
  const int L = ld[b];
  amh[idx] = (r < L && c < L) ? (_Float16)am[idx] : (_Float16)0.f;
}

// ---------------------------------------------------------------------------
// Transpose the 4 LSTM weight matrices [256,64] -> [64,256]
// ---------------------------------------------------------------------------
__global__ void k_twt(const float* __restrict__ a, const float* __restrict__ b,
                      const float* __restrict__ c, const float* __restrict__ d,
                      float* __restrict__ WT)
{
  const int mat = blockIdx.x, k = blockIdx.y, g = threadIdx.x;
  const float* src = (mat == 0) ? a : (mat == 1) ? b : (mat == 2) ? c : d;
  WT[((long)mat * 64 + k) * 256 + g] = src[(long)g * 64 + k];
}

// ---------------------------------------------------------------------------
// LSTM, one block per batch element, 256 threads (one per gate row).
// Torch gate order i,f,g,o.  Mask t<L holds h/c past sequence end.
// ---------------------------------------------------------------------------
__global__ __launch_bounds__(256)
void k_lstm(const float* __restrict__ x, const int* __restrict__ ld,
            const float* __restrict__ WihT, const float* __restrict__ WhhT,
            const float* __restrict__ bih, const float* __restrict__ bhh,
            float* __restrict__ out, int out_stride, int backward)
{
  __shared__ float hbuf[64], cbuf[64], gates[256], xs[64];
  const int b = blockIdx.x, g = threadIdx.x;
  const int L = ld[b];
  if (g < 64) { hbuf[g] = 0.f; cbuf[g] = 0.f; }
  const float bsum = bih[g] + bhh[g];
  __syncthreads();

  for (int t = 0; t < 512; ++t) {
    const int idx = backward ? ((t < L) ? (L - 1 - t) : 0) : t;
    if (g < 64) xs[g] = x[((long)b * 512 + idx) * 64 + g];
    __syncthreads();

    float acc = bsum;
    for (int k = 0; k < 64; ++k)
      acc += xs[k] * WihT[k * 256 + g] + hbuf[k] * WhhT[k * 256 + g];
    gates[g] = acc;
    __syncthreads();

    if (g < 64) {
      const float ig = 1.f / (1.f + __expf(-gates[g]));
      const float fg = 1.f / (1.f + __expf(-gates[64 + g]));
      const float gg = tanhf(gates[128 + g]);
      const float og = 1.f / (1.f + __expf(-gates[192 + g]));
      const float c  = fg * cbuf[g] + ig * gg;
      const float hn = og * tanhf(c);
      if (t < L) { cbuf[g] = c; hbuf[g] = hn; }
      out[((long)b * 512 + t) * out_stride + g] = hbuf[g];
    }
    __syncthreads();
  }
}

// hcat[:, :, 64:128] = flip(backward-lstm output):  out[t] = ys[clip(L-1-t,0,511)]
__global__ void k_bfix(const float* __restrict__ ysb, const int* __restrict__ ld,
                       float* __restrict__ hcat)
{
  const int t = blockIdx.x, b = blockIdx.y, u = threadIdx.x;  // grid(512,B), blk 64
  const int L = ld[b];
  int src = L - 1 - t;
  src = src < 0 ? 0 : (src > 511 ? 511 : src);
  hcat[((long)b * 512 + t) * 128 + 64 + u] = ysb[((long)b * 512 + src) * 64 + u];
}

// Masked max-pool over valid positions
__global__ void k_pool(const float* __restrict__ x, const int* __restrict__ ld,
                       float* __restrict__ pooled)
{
  const int b = blockIdx.x;
  const int L = ld[b];
  for (int c = threadIdx.x; c < 512; c += blockDim.x) {
    float m = -3.402823e38f;
    for (int i = 0; i < L; ++i)
      m = fmaxf(m, x[((long)b * 512 + i) * 512 + c]);
    pooled[(long)b * 512 + c] = m;
  }
}

// Small dense layer (rows = B = 32)
__global__ void k_fc(const float* __restrict__ x, const float* __restrict__ W,
                     const float* __restrict__ bias,
                     const float* __restrict__ gam, const float* __restrict__ bet,
                     const float* __restrict__ mea, const float* __restrict__ var,
                     float* __restrict__ y, int N, int K, int act)
{
  const int b = blockIdx.x;
  const int n = blockIdx.y * blockDim.x + threadIdx.x;
  if (n >= N) return;
  float acc = 0.f;
  const float* xb = x + (long)b * K;
  const float* wr = W + (long)n * K;
  for (int k = 0; k < K; ++k) acc += xb[k] * wr[k];
  if (gam) {
    const float s = gam[n] * rsqrtf(var[n] + BN_EPS);
    acc = (acc + bias[n] - mea[n]) * s + bet[n];
  } else {
    acc += bias[n];
  }
  if (act == 2) acc = acc >= 0.f ? acc : 0.01f * acc;
  y[(long)b * N + n] = acc;
}

// ---------------------------------------------------------------------------
extern "C" void kernel_launch(void* const* d_in, const int* in_sizes, int n_in,
                              void* d_out, int out_size, void* d_ws, size_t ws_size,
                              hipStream_t stream)
{
  (void)in_sizes; (void)n_in; (void)out_size; (void)ws_size;
  const float* x_o = (const float*)d_in[0];
  const int*   ld  = (const int*)d_in[1];
  const float* am  = (const float*)d_in[2];
#define FIN_P(i) ((const float*)d_in[(i)])
  // mlp1: 3..8  lstm_f: 9..12  lstm_b: 13..16  mlp2: 17..22  gcl: 23..28
  // g1a: 29..34 g1b: 35..40  g2res: 41..46  g2a: 47..52  g2b: 53..58
  // fc1: 59..64 fc2: 65,66

  char* ws = (char*)d_ws;
  size_t off = 0;
  auto alloc = [&](size_t bytes) -> char* {
    char* p = ws + off;
    off += (bytes + 255) & ~(size_t)255;
    return p;
  };
  const long ML = 32L * 512;                                  // 16384 rows
  _Float16* amh = (_Float16*)alloc(32L * 512 * 512 * 2);
  float* WT     = (float*)alloc(4L * 64 * 256 * 4);
  float* h1     = (float*)alloc(ML * 64 * 4);
  float* hcat   = (float*)alloc(ML * 128 * 4);
  float* ysb    = (float*)alloc(ML * 64 * 4);
  float* t256   = (float*)alloc(ML * 256 * 4);
  float* b0     = (float*)alloc(ML * 256 * 4);
  float* b1     = (float*)alloc(ML * 256 * 4);
  float* b2     = (float*)alloc(ML * 256 * 4);
  float* r512   = (float*)alloc(ML * 512 * 4);
  float* x7     = (float*)alloc(ML * 512 * 4);
  float* pooled = (float*)alloc(32L * 512 * 4);
  float* y1     = (float*)alloc(32L * 512 * 4);

  k_twt<<<dim3(4, 64), 256, 0, stream>>>(FIN_P(9), FIN_P(10), FIN_P(13), FIN_P(14), WT);
  k_amh<<<32768, 256, 0, stream>>>(am, ld, amh);

  // mlp1: [*,32] -> [*,64], lrelu
  k_gemm_lin<<<dim3(256, 1), 128, 0, stream>>>(x_o, FIN_P(3), FIN_P(4),
      FIN_P(5), FIN_P(6), FIN_P(7), FIN_P(8), nullptr, h1, 16384, 64, 32, 2);

  // bi-LSTM
  k_lstm<<<32, 256, 0, stream>>>(h1, ld, WT, WT + 64 * 256,
      FIN_P(11), FIN_P(12), hcat, 128, 0);
  k_lstm<<<32, 256, 0, stream>>>(h1, ld, WT + 2 * 64 * 256, WT + 3 * 64 * 256,
      FIN_P(15), FIN_P(16), ysb, 64, 1);
  k_bfix<<<dim3(512, 32), 64, 0, stream>>>(ysb, ld, hcat);

  // mlp2: [*,128] -> [*,256], lrelu -> b0
  k_gemm_lin<<<dim3(256, 4), 128, 0, stream>>>(hcat, FIN_P(17), FIN_P(18),
      FIN_P(19), FIN_P(20), FIN_P(21), FIN_P(22), nullptr, b0, 16384, 256, 128, 2);

  // gcl -> b1 (residual r1)
  k_gemm_agg<<<dim3(8, 4, 32), 128, 0, stream>>>(amh, b0, t256, 256);
  k_gemm_lin<<<dim3(256, 4), 128, 0, stream>>>(t256, FIN_P(23), FIN_P(24),
      FIN_P(25), FIN_P(26), FIN_P(27), FIN_P(28), nullptr, b1, 16384, 256, 256, 2);

  // g1a -> b2
  k_gemm_agg<<<dim3(8, 4, 32), 128, 0, stream>>>(amh, b1, t256, 256);
  k_gemm_lin<<<dim3(256, 4), 128, 0, stream>>>(t256, FIN_P(29), FIN_P(30),
      FIN_P(31), FIN_P(32), FIN_P(33), FIN_P(34), nullptr, b2, 16384, 256, 256, 1);

  // g1b (+res b1) -> b0
  k_gemm_agg<<<dim3(8, 4, 32), 128, 0, stream>>>(amh, b2, t256, 256);
  k_gemm_lin<<<dim3(256, 4), 128, 0, stream>>>(t256, FIN_P(35), FIN_P(36),
      FIN_P(37), FIN_P(38), FIN_P(39), FIN_P(40), b1, b0, 16384, 256, 256, 2);

  // g2res: lbn(b0) -> r512
  k_gemm_lin<<<dim3(256, 8), 128, 0, stream>>>(b0, FIN_P(41), FIN_P(42),
      FIN_P(43), FIN_P(44), FIN_P(45), FIN_P(46), nullptr, r512, 16384, 512, 256, 0);

  // g2a -> b1
  k_gemm_agg<<<dim3(8, 4, 32), 128, 0, stream>>>(amh, b0, t256, 256);
  k_gemm_lin<<<dim3(256, 4), 128, 0, stream>>>(t256, FIN_P(47), FIN_P(48),
      FIN_P(49), FIN_P(50), FIN_P(51), FIN_P(52), nullptr, b1, 16384, 256, 256, 1);

  // g2b (+res r512) -> x7 [*,512]
  k_gemm_agg<<<dim3(8, 4, 32), 128, 0, stream>>>(amh, b1, t256, 256);
  k_gemm_lin<<<dim3(256, 8), 128, 0, stream>>>(t256, FIN_P(53), FIN_P(54),
      FIN_P(55), FIN_P(56), FIN_P(57), FIN_P(58), r512, x7, 16384, 512, 256, 2);

  // pool + fc1 + fc2
  k_pool<<<32, 256, 0, stream>>>(x7, ld, pooled);
  k_fc<<<dim3(32, 4), 128, 0, stream>>>(pooled, FIN_P(59), FIN_P(60),
      FIN_P(61), FIN_P(62), FIN_P(63), FIN_P(64), y1, 512, 512, 2);
  k_fc<<<dim3(32, 4), 128, 0, stream>>>(y1, FIN_P(65), FIN_P(66),
      nullptr, nullptr, nullptr, nullptr, (float*)d_out, 400, 512, 0);
#undef FIN_P
}
